// Text2Emoji_65833258713137
// MI455X (gfx1250) — matrix operand order repaired and checked by reading, not compiled, not run on hardware
//
#include <hip/hip_runtime.h>
#include <math.h>

// ---------------------------------------------------------------------------
// Types for CDNA5 WMMA (wave32): v_wmma_f32_16x16x32_bf16
// ---------------------------------------------------------------------------
typedef __bf16 bf16_t;
typedef bf16_t v8bf  __attribute__((ext_vector_type(8)));
typedef bf16_t v16bf __attribute__((ext_vector_type(16)));
typedef float  v8f   __attribute__((ext_vector_type(8)));

#define CAT16(x0, x1) __builtin_shufflevector(x0, x1, 0,1,2,3,4,5,6,7,8,9,10,11,12,13,14,15)

// ---------------------------------------------------------------------------
// Generic WMMA GEMM:  C[M,N] = A[M,K](bf16) @ W[N,K](bf16)^T  (+ bias[N])
// One wave -> one 16x64 tile: 4 independent accumulator chains share the A
// operand. All 10 loads of a K-step are issued before any WMMA (distinct
// named locals -> distinct VGPRs -> partial s_wait_loadcnt instead of full).
// Requires M%16==0, N%64==0, K%32==0.  ntstore=1 -> non-temporal C stores.
// ---------------------------------------------------------------------------
__global__ __launch_bounds__(256) void k_gemm_bf16(
    const bf16_t* __restrict__ A, int lda,
    const bf16_t* __restrict__ W, int ldw,
    const float* __restrict__ bias,
    float* __restrict__ C, int ldc,
    int M, int N, int K, int ntstore)
{
  int wave = blockIdx.x * (blockDim.x >> 5) + (threadIdx.x >> 5);
  int lane = threadIdx.x & 31;
  int tilesN = N >> 6;                       // 64-wide tiles
  int total  = (M >> 4) * tilesN;
  if (wave >= total) return;                 // wave-uniform: EXEC stays all-1
  int tn = wave % tilesN;
  int tm = wave / tilesN;
  int l15 = lane & 15;
  int hi  = lane >> 4;                       // lane group 0..15 vs 16..31

  // A operand: lane m holds A[m][k..k+15] (hi lanes: k+16..k+31), contiguous
  const bf16_t* Ap  = A + (size_t)((tm << 4) + l15) * lda + hi * 16;
  // B operand: lane n holds W[n][k..k+15] (hi lanes: k+16..k+31), contiguous
  const bf16_t* Wp0 = W + (size_t)((tn << 6) + l15) * ldw + hi * 16;
  const bf16_t* Wp1 = Wp0 + (size_t)16 * ldw;
  const bf16_t* Wp2 = Wp0 + (size_t)32 * ldw;
  const bf16_t* Wp3 = Wp0 + (size_t)48 * ldw;

  v8f acc0 = (v8f){0.f,0.f,0.f,0.f,0.f,0.f,0.f,0.f};
  v8f acc1 = acc0, acc2 = acc0, acc3 = acc0;

  for (int k = 0; k < K; k += 32) {
    // issue all loads first (one clause; distinct destination VGPRs)
    v8bf a0  = *(const v8bf*)(Ap  + k);
    v8bf a1  = *(const v8bf*)(Ap  + k + 8);
    v8bf b00 = *(const v8bf*)(Wp0 + k);
    v8bf b01 = *(const v8bf*)(Wp0 + k + 8);
    v8bf b10 = *(const v8bf*)(Wp1 + k);
    v8bf b11 = *(const v8bf*)(Wp1 + k + 8);
    v8bf b20 = *(const v8bf*)(Wp2 + k);
    v8bf b21 = *(const v8bf*)(Wp2 + k + 8);
    v8bf b30 = *(const v8bf*)(Wp3 + k);
    v8bf b31 = *(const v8bf*)(Wp3 + k + 8);

    v16bf av  = CAT16(a0, a1);
    v16bf bv0 = CAT16(b00, b01);
    v16bf bv1 = CAT16(b10, b11);
    v16bf bv2 = CAT16(b20, b21);
    v16bf bv3 = CAT16(b30, b31);

    acc0 = __builtin_amdgcn_wmma_f32_16x16x32_bf16(false, av, false, bv0, (short)0, acc0, false, false);
    acc1 = __builtin_amdgcn_wmma_f32_16x16x32_bf16(false, av, false, bv1, (short)0, acc1, false, false);
    acc2 = __builtin_amdgcn_wmma_f32_16x16x32_bf16(false, av, false, bv2, (short)0, acc2, false, false);
    acc3 = __builtin_amdgcn_wmma_f32_16x16x32_bf16(false, av, false, bv3, (short)0, acc3, false, false);
  }

  int mb = (tm << 4) + (hi << 3);            // VGPR i -> M = mb + i
  int n0 = (tn << 6) + l15;
  float bb0 = bias ? bias[n0]      : 0.f;
  float bb1 = bias ? bias[n0 + 16] : 0.f;
  float bb2 = bias ? bias[n0 + 32] : 0.f;
  float bb3 = bias ? bias[n0 + 48] : 0.f;
  if (ntstore) {
#pragma unroll
    for (int i = 0; i < 8; i++) {
      float* row = C + (size_t)(mb + i) * ldc + n0;
      __builtin_nontemporal_store(acc0[i] + bb0, row);
      __builtin_nontemporal_store(acc1[i] + bb1, row + 16);
      __builtin_nontemporal_store(acc2[i] + bb2, row + 32);
      __builtin_nontemporal_store(acc3[i] + bb3, row + 48);
    }
  } else {
#pragma unroll
    for (int i = 0; i < 8; i++) {
      float* row = C + (size_t)(mb + i) * ldc + n0;
      row[0]  = acc0[i] + bb0;
      row[16] = acc1[i] + bb1;
      row[32] = acc2[i] + bb2;
      row[48] = acc3[i] + bb3;
    }
  }
}

// ---------------------------------------------------------------------------
// Elementwise helpers
// ---------------------------------------------------------------------------
__global__ void k_cvt(const float* __restrict__ x, bf16_t* __restrict__ y, int n) {
  int i = blockIdx.x * blockDim.x + threadIdx.x;
  if (i < n) y[i] = (bf16_t)x[i];
}

__global__ void k_init_h(float* __restrict__ h, bf16_t* __restrict__ hb, int n) {
  int i = blockIdx.x * blockDim.x + threadIdx.x;
  if (i < n) { h[i] = 0.f; hb[i] = (bf16_t)0.f; }
}

// Embedding gather -> bf16 rows. Row i covers (r = i/64, c = i%64).
// swap=1 (encoder): token = tok[c*64 + r]  (reference's transpose(1,0,2) quirk)
// swap=0 (decoder): token = tok[i]
__global__ void k_embed(const int* __restrict__ tok, const float* __restrict__ emb,
                        bf16_t* __restrict__ out, int swap) {
  int i = blockIdx.x;
  int r = i >> 6, c = i & 63;
  int t = swap ? tok[c * 64 + r] : tok[i];
  const float* e = emb + (size_t)t * 256;
  out[(size_t)i * 256 + threadIdx.x] = (bf16_t)e[threadIdx.x];
}

// GRU gate math: gi/gh are (64,768) rows (bias already folded in by GEMM).
// grid 64, block 256 -> idx in [0, 64*256)
__global__ void k_gru_gate(const float* __restrict__ gi, const float* __restrict__ gh,
                           const float* __restrict__ h_prev,
                           float* __restrict__ h_new, bf16_t* __restrict__ h_new_bf,
                           bf16_t* __restrict__ ys, int ys_stride) {
  int idx = blockIdx.x * blockDim.x + threadIdx.x;
  int b = idx >> 8, j = idx & 255;
  const float* gib = gi + b * 768;
  const float* ghb = gh + b * 768;
  float r = 1.f / (1.f + expf(-(gib[j] + ghb[j])));
  float z = 1.f / (1.f + expf(-(gib[256 + j] + ghb[256 + j])));
  float nn = tanhf(gib[512 + j] + r * ghb[512 + j]);
  float hp = h_prev[idx];
  float h  = (1.f - z) * nn + z * hp;
  h_new[idx]    = h;
  h_new_bf[idx] = (bf16_t)h;
  if (ys) ys[b * ys_stride + j] = (bf16_t)h;
}

// state0[b] = enc_seq[b, len_b] where len_b = (#nonzero tokens in column b) - 1
__global__ void k_state0(const int* __restrict__ src, const float* __restrict__ enc_seq,
                         float* __restrict__ state, bf16_t* __restrict__ state_bf) {
  const int S = 64, B = 64, H = 256;
  int b = blockIdx.x, j = threadIdx.x;
  int len = 0;
  for (int s = 0; s < S; s++) len += (src[s * B + b] != 0);
  len -= 1;
  float v = enc_seq[((size_t)b * S + len) * H + j];
  state[b * H + j] = v;
  state_bf[b * H + j] = (bf16_t)v;
}

// Additive attention + state update: grid B=64, block S=64
// sc[s] = w1 . tanh(enc_part[b,s,:] + h_part[b,:]) + b1 ; masked softmax ;
// ctx = sum_s w[s] * enc_seq[b,s,:] ; state = h + ctx
__global__ void k_attention(const int* __restrict__ src,
                            const float* __restrict__ enc_seq,
                            const float* __restrict__ enc_part,
                            const float* __restrict__ h_part,
                            const float* __restrict__ att_w1,
                            const float* __restrict__ att_b1,
                            const float* __restrict__ h_cur,
                            float* __restrict__ state, bf16_t* __restrict__ state_bf) {
  const int S = 64, B = 64, H = 256;
  __shared__ float hp[256];
  __shared__ float wsc[64];
  __shared__ float red[64];
  int b = blockIdx.x, s = threadIdx.x;
  for (int i = s; i < H; i += 64) hp[i] = h_part[b * H + i];
  __syncthreads();

  const float* ep = enc_part + ((size_t)b * S + s) * H;
  float acc = 0.f;
  for (int k = 0; k < H; k++) acc += att_w1[k] * tanhf(ep[k] + hp[k]);
  acc += att_b1[0];
  bool valid = (src[s * B + b] != 0);
  float scv = valid ? acc : -3.0e38f;

  red[s] = scv; __syncthreads();
  for (int o = 32; o > 0; o >>= 1) { if (s < o) red[s] = fmaxf(red[s], red[s + o]); __syncthreads(); }
  float mx = red[0]; __syncthreads();

  float e = valid ? expf(scv - mx) : 0.f;
  red[s] = e; __syncthreads();
  for (int o = 32; o > 0; o >>= 1) { if (s < o) red[s] += red[s + o]; __syncthreads(); }
  float inv = 1.f / red[0];
  wsc[s] = e * inv; __syncthreads();

  for (int j = s; j < H; j += 64) {
    float c = 0.f;
    for (int s2 = 0; s2 < S; s2++) c += wsc[s2] * enc_seq[((size_t)b * S + s2) * H + j];
    float st = h_cur[b * H + j] + c;
    state[b * H + j] = st;
    state_bf[b * H + j] = (bf16_t)st;
  }
}

// ---------------------------------------------------------------------------
// Host orchestration
// ---------------------------------------------------------------------------
static inline int cdiv(int a, int b) { return (a + b - 1) / b; }

static void gemm(hipStream_t st, const bf16_t* A, int lda, const bf16_t* W, int ldw,
                 const float* bias, float* C, int ldc, int M, int N, int K,
                 int nt = 0) {
  int tiles = (M / 16) * (N / 64);
  k_gemm_bf16<<<cdiv(tiles, 8), 256, 0, st>>>(A, lda, W, ldw, bias, C, ldc, M, N, K, nt);
}

static void cvt(hipStream_t st, const float* x, bf16_t* y, int n) {
  k_cvt<<<cdiv(n, 256), 256, 0, st>>>(x, y, n);
}

extern "C" void kernel_launch(void* const* d_in, const int* in_sizes, int n_in,
                              void* d_out, int out_size, void* d_ws, size_t ws_size,
                              hipStream_t stream) {
  (void)in_sizes; (void)n_in; (void)out_size; (void)ws_size;
  const int S = 64, B = 64, Tm1 = 31, H = 256, VD = 32000;
  const int SB = S * B;            // 4096 rows in recurrence layout
  const int DB = Tm1 * B;          // 1984 decoder rows

  // ---- inputs (setup_inputs dict order; enc_gru pytree-flattened) ----
  const int*   src      = (const int*)d_in[0];     // (S,B)
  const int*   tgt      = (const int*)d_in[1];     // (T,B)
  const float* enc_emb  = (const float*)d_in[2];   // (VE,E)
  // d_in[3..18]: enc_gru[l][d] = (wih, whh, bih, bhh)
  const float* hid_w    = (const float*)d_in[19];  // (H, 2H)
  const float* hid_b    = (const float*)d_in[20];  // (H,)
  const float* dec_emb  = (const float*)d_in[21];  // (VD,E)
  const float* cell_wih = (const float*)d_in[22];  // (3H,E)
  const float* cell_whh = (const float*)d_in[23];  // (3H,H)
  const float* cell_bih = (const float*)d_in[24];
  const float* cell_bhh = (const float*)d_in[25];
  const float* out_w    = (const float*)d_in[26];  // (VD,H)
  const float* out_b    = (const float*)d_in[27];
  const float* att_w0   = (const float*)d_in[28];  // (H,2H)
  const float* att_b0   = (const float*)d_in[29];
  const float* att_w1   = (const float*)d_in[30];  // (H,)
  const float* att_b1   = (const float*)d_in[31];  // ()
  float* out = (float*)d_out;                      // (B, T-1, VD)

  // ---- bump allocator on workspace ----
  char* ws = (char*)d_ws;
  size_t off = 0;
  auto alloc = [&](size_t bytes) -> void* {
    void* p = ws + off;
    off += (bytes + 255) & ~(size_t)255;
    return p;
  };

  // bf16 weight copies
  bf16_t* w_ih[2][2]; bf16_t* w_hh[2][2];
  const float* f_ih[2][2]; const float* f_hh[2][2];
  const float* bih[2][2]; const float* bhh[2][2];
  for (int l = 0; l < 2; l++)
    for (int d = 0; d < 2; d++) {
      int base = 3 + l * 8 + d * 4;
      f_ih[l][d] = (const float*)d_in[base + 0];
      f_hh[l][d] = (const float*)d_in[base + 1];
      bih[l][d]  = (const float*)d_in[base + 2];
      bhh[l][d]  = (const float*)d_in[base + 3];
      int in_dim = (l == 0) ? 256 : 512;
      w_ih[l][d] = (bf16_t*)alloc((size_t)768 * in_dim * 2);
      w_hh[l][d] = (bf16_t*)alloc((size_t)768 * 256 * 2);
    }
  bf16_t* w_hid  = (bf16_t*)alloc((size_t)256 * 512 * 2);
  bf16_t* w_cih  = (bf16_t*)alloc((size_t)768 * 256 * 2);
  bf16_t* w_chh  = (bf16_t*)alloc((size_t)768 * 256 * 2);
  bf16_t* w_out  = (bf16_t*)alloc((size_t)VD * 256 * 2);
  bf16_t* w_att0 = (bf16_t*)alloc((size_t)256 * 512 * 2);

  // activations / scratch
  bf16_t* xs_a     = (bf16_t*)alloc((size_t)SB * 512 * 2);
  bf16_t* xs_b     = (bf16_t*)alloc((size_t)SB * 512 * 2);
  float*  gi       = (float*)alloc((size_t)SB * 768 * 4);
  float*  gh       = (float*)alloc((size_t)B * 768 * 4);
  float*  hbuf     = (float*)alloc((size_t)B * H * 4);
  bf16_t* hbf      = (bf16_t*)alloc((size_t)B * H * 2);
  float*  enc_seq  = (float*)alloc((size_t)SB * H * 4);
  bf16_t* enc_sbf  = (bf16_t*)alloc((size_t)SB * H * 2);
  float*  enc_part = (float*)alloc((size_t)SB * H * 4);
  bf16_t* xd       = (bf16_t*)alloc((size_t)DB * 256 * 2);
  float*  gi_dec   = (float*)alloc((size_t)DB * 768 * 4);
  float*  hdec     = (float*)alloc((size_t)B * H * 4);
  bf16_t* hdec_bf  = (bf16_t*)alloc((size_t)B * H * 2);
  float*  hpart    = (float*)alloc((size_t)B * H * 4);
  float*  state    = (float*)alloc((size_t)B * H * 4);
  bf16_t* state_bf = (bf16_t*)alloc((size_t)B * H * 2);

  // ---- weight conversions (f32 -> bf16) ----
  for (int l = 0; l < 2; l++)
    for (int d = 0; d < 2; d++) {
      int in_dim = (l == 0) ? 256 : 512;
      cvt(stream, f_ih[l][d], w_ih[l][d], 768 * in_dim);
      cvt(stream, f_hh[l][d], w_hh[l][d], 768 * 256);
    }
  cvt(stream, hid_w,    w_hid,  256 * 512);
  cvt(stream, cell_wih, w_cih,  768 * 256);
  cvt(stream, cell_whh, w_chh,  768 * 256);
  cvt(stream, out_w,    w_out,  VD * 256);
  cvt(stream, att_w0,   w_att0, 256 * 512);

  // ---- encoder ----
  // Recurrence layout: row (t*64 + j) with t = original batch index b,
  // j = original sequence index s  (matches reference's transpose+scan).
  k_embed<<<SB, 256, 0, stream>>>(src, enc_emb, xs_a, /*swap=*/1);

  bf16_t* xin = xs_a; int lin = 256;
  bf16_t* xout = xs_b;
  for (int l = 0; l < 2; l++) {
    for (int d = 0; d < 2; d++) {
      // input-side gates for all timesteps at once
      gemm(stream, xin, lin, w_ih[l][d], lin, bih[l][d], gi, 768, SB, 768, lin);
      k_init_h<<<cdiv(B * H, 256), 256, 0, stream>>>(hbuf, hbf, B * H);
      for (int step = 0; step < 64; step++) {
        int t = d ? (63 - step) : step;
        gemm(stream, hbf, 256, w_hh[l][d], 256, bhh[l][d], gh, 768, B, 768, 256);
        bf16_t* ys = xout + (size_t)t * 64 * 512 + d * 256;
        k_gru_gate<<<64, 256, 0, stream>>>(gi + (size_t)t * 64 * 768, gh, hbuf,
                                           hbuf, hbf, ys, 512);
      }
    }
    bf16_t* tmp = xin; xin = xout; xout = tmp;
    lin = 512;
  }
  // xin: final encoder output (4096, 512) bf16, rows ordered (b*64 + s)

  // enc_seq (B,S,H) = enc_out @ hid_w^T + hid_b   (row order identical)
  gemm(stream, xin, 512, w_hid, 512, hid_b, enc_seq, 256, SB, 256, 512);
  cvt(stream, enc_seq, enc_sbf, SB * H);
  // enc_part = enc_seq @ att_w0[:, :H]^T   (bias folded into h_part later)
  gemm(stream, enc_sbf, 256, w_att0, 512, nullptr, enc_part, 256, SB, 256, 256);
  k_state0<<<B, 256, 0, stream>>>(src, enc_seq, state, state_bf);

  // ---- decoder ----
  k_embed<<<DB, 256, 0, stream>>>(tgt, dec_emb, xd, /*swap=*/0);
  gemm(stream, xd, 256, w_cih, 256, cell_bih, gi_dec, 768, DB, 768, 256);

  for (int t = 0; t < Tm1; t++) {
    // gh = state @ cell_whh^T + cell_bhh
    gemm(stream, state_bf, 256, w_chh, 256, cell_bhh, gh, 768, B, 768, 256);
    // h_t
    k_gru_gate<<<64, 256, 0, stream>>>(gi_dec + (size_t)t * 64 * 768, gh, state,
                                       hdec, hdec_bf, nullptr, 0);
    // logits -> d_out[(b*31 + t)*VD + v]  (non-temporal: never re-read,
    // keeps out_w resident in the 192MB L2 across all 31 steps)
    gemm(stream, hdec_bf, 256, w_out, 256, out_b, out + (size_t)t * VD,
         Tm1 * VD, B, VD, 256, /*nt=*/1);
    // h_part = h @ att_w0[:, H:]^T + att_b0
    gemm(stream, hdec_bf, 256, w_att0 + 256, 512, att_b0, hpart, 256, B, 256, 256);
    // attention + state update
    k_attention<<<B, 64, 0, stream>>>(src, enc_seq, enc_part, hpart,
                                      att_w1, att_b1, hdec, state, state_bf);
  }
}
